// GFA_72464688218295
// MI455X (gfx1250) — matrix operand order repaired and checked by reading, not compiled, not run on hardware
//
#include <hip/hip_runtime.h>

// B=4, C=64, N=8192 rank-1 attention, flash-style with WMMA f16 and
// double-buffered LDS staging (async-to-LDS when the toolchain exposes it).
#define BB   4
#define CC   64
#define NN   8192
#define CEXT 80            // 64 channels + ones-channel(64) + 15 zero pad -> 5 WMMA tiles
#define EPSB 1e-5f
#define LOG2E 1.4426950408889634f

// LDS staging geometry (per 128-wide m superstep)
#define MSTEP   128
#define ROWB    272                    // 128 halves = 256B, padded to 17*16B (bank-conflict-free)
#define FBYTES  (CEXT * ROWB)          // 21760 B of fext tile
#define BOFFB   FBYTES                 // b[] row (128 f32 = 512 B) follows fext
#define STAGEB  (FBYTES + 512)         // 22272 B per stage (16B multiple)
#define FCHUNKS (CEXT * 16)            // 1280 16B chunks of fext per stage
#define NCHUNKS (FCHUNKS + 32)         // + 32 chunks of b

typedef __attribute__((ext_vector_type(16))) _Float16 v16h;
typedef __attribute__((ext_vector_type(8)))  _Float16 v8h;
typedef __attribute__((ext_vector_type(8)))  float    v8f;
typedef __attribute__((ext_vector_type(4)))  int      v4i;
typedef _Float16 half_t;

// Probe-derived pointer types for the async-to-LDS builtin:
//   arg0: v4i addrspace(1)*  (printed as "int __vector(4) __device__ *")
//   arg1: v4i addrspace(3)*  (printed as "int __vector(4) __shared__ *")
typedef v4i __attribute__((address_space(1)))* gv4i_p;
typedef v4i __attribute__((address_space(3)))* sv4i_p;

#if defined(__HIP_DEVICE_COMPILE__) && defined(__gfx1250__) && \
    __has_builtin(__builtin_amdgcn_global_load_async_to_lds_b128)
#define USE_ASYNC_LDS 1
#endif

// ---------------------------------------------------------------- a,b vectors
__global__ void ab_kernel(const float* __restrict__ x,
                          const float* __restrict__ wa, const float* __restrict__ ba,
                          const float* __restrict__ wb, const float* __restrict__ bb,
                          float* __restrict__ av, float* __restrict__ bv) {
    int b = blockIdx.y;
    int n = blockIdx.x * blockDim.x + threadIdx.x;
    const float* xp = x + (size_t)b * CC * NN + n;
    float va = 0.f, vb = 0.f;
#pragma unroll 8
    for (int c = 0; c < CC; ++c) {
        float xv = xp[(size_t)c * NN];        // coalesced: lanes stride n
        va = fmaf(xv, wa[c], va);             // wa[c] uniform -> s_load
        vb = fmaf(xv, wb[c], vb);
    }
    av[b * NN + n] = va + ba[0];
    bv[b * NN + n] = vb + bb[0];
}

// ------------------------------------------------------- per-batch min/max(b)
__global__ void bminmax_kernel(const float* __restrict__ bv, float* __restrict__ bmm) {
    __shared__ float smin[256], smax[256];
    int b = blockIdx.x;
    float mn = 3.4e38f, mx = -3.4e38f;
    for (int i = threadIdx.x; i < NN; i += 256) {
        float v = bv[b * NN + i];
        mn = fminf(mn, v);
        mx = fmaxf(mx, v);
    }
    smin[threadIdx.x] = mn; smax[threadIdx.x] = mx;
    __syncthreads();
    for (int s = 128; s > 0; s >>= 1) {
        if (threadIdx.x < (unsigned)s) {
            smin[threadIdx.x] = fminf(smin[threadIdx.x], smin[threadIdx.x + s]);
            smax[threadIdx.x] = fmaxf(smax[threadIdx.x], smax[threadIdx.x + s]);
        }
        __syncthreads();
    }
    if (threadIdx.x == 0) { bmm[b * 2 + 0] = smin[0]; bmm[b * 2 + 1] = smax[0]; }
}

// ------------------- fext[b][d][n] f16: BN-folded mlp GEMM + ones/zeros tiles
__global__ void fext_kernel(const float* __restrict__ x,  const float* __restrict__ wm,
                            const float* __restrict__ bm, const float* __restrict__ gamma,
                            const float* __restrict__ beta, const float* __restrict__ rmean,
                            const float* __restrict__ rvar, half_t* __restrict__ fext) {
    const int blocksPerBatch = NN / 128;
    int b = blockIdx.x / blocksPerBatch;
    int n = (blockIdx.x % blocksPerBatch) * 128 + threadIdx.x;
    float xr[CC];
#pragma unroll
    for (int c = 0; c < CC; ++c)
        xr[c] = x[((size_t)(b * CC + c)) * NN + n];   // coalesced per c
    for (int d = 0; d < CC; ++d) {
        float scale = gamma[d] * rsqrtf(rvar[d] + EPSB);
        float bias  = (bm[d] - rmean[d]) * scale + beta[d];
        float acc = 0.f;
#pragma unroll
        for (int c = 0; c < CC; ++c)
            acc = fmaf(xr[c], wm[d * CC + c], acc);   // wm uniform -> s_load
        fext[((size_t)(b * CEXT + d)) * NN + n] = (half_t)(acc * scale + bias);
    }
    fext[((size_t)(b * CEXT + CC)) * NN + n] = (half_t)1.0f;   // row-sum channel
#pragma unroll
    for (int d = CC + 1; d < CEXT; ++d)
        fext[((size_t)(b * CEXT + d)) * NN + n] = (half_t)0.0f;
}

// ---------------------------------------------------------- stage copy helpers
// Copies one 128-wide m superstep of fext (80 rows, padded) + b[ms..ms+128)
// into LDS stage `st`. 1312 16-byte chunks spread over 128 threads.
__device__ __forceinline__ void copy_stage(char* smem, int st,
                                           const half_t* __restrict__ fb,
                                           const float* __restrict__ bvb, int ms) {
    char* sdst = smem + (size_t)st * STAGEB;
#pragma unroll
    for (int i = 0; i < 11; ++i) {
        int idx = threadIdx.x + 128 * i;
        if (idx < NCHUNKS) {
            const char* gsrc;
            int loff;
            if (idx < FCHUNKS) {
                int c = idx >> 4, j = idx & 15;
                gsrc = (const char*)fb + (size_t)c * (NN * 2) + (size_t)ms * 2 + j * 16;
                loff = c * ROWB + j * 16;
            } else {
                int j = idx - FCHUNKS;
                gsrc = (const char*)(bvb + ms) + j * 16;
                loff = BOFFB + j * 16;
            }
#if defined(USE_ASYNC_LDS)
            __builtin_amdgcn_global_load_async_to_lds_b128(
                (gv4i_p)gsrc, (sv4i_p)(sdst + loff), 0, 0);
#else
            *(float4*)(sdst + loff) = *(const float4*)gsrc;
#endif
        }
    }
}

__device__ __forceinline__ void stage_wait() {
#if defined(USE_ASYNC_LDS)
#if __has_builtin(__builtin_amdgcn_s_wait_asynccnt)
    __builtin_amdgcn_s_wait_asynccnt(0);
#else
    asm volatile("s_wait_asynccnt 0x0" ::: "memory");
#endif
#endif
}

// -------------------------------------------------- main flash-style WMMA pass
__global__ void __launch_bounds__(128)
attn_kernel(const float* __restrict__ feature, const float* __restrict__ av,
            const float* __restrict__ bv, const float* __restrict__ bmm,
            const half_t* __restrict__ fext, float* __restrict__ out) {
    __shared__ __align__(16) char smem[2 * STAGEB];     // 43.5 KB double buffer

    const int tilesPerBatch = NN / 64;
    int b     = blockIdx.x / tilesPerBatch;
    int chunk = blockIdx.x % tilesPerBatch;

    int wave = threadIdx.x >> 5;
    int lane = threadIdx.x & 31;
    int hi   = lane >> 4;                          // lane half (A/B layout split)
    int col  = lane & 15;                          // row M (A) / column N (B,C,D)
    int n0   = chunk * 64 + wave * 16;             // this wave's 16 query rows

    // log2-domain a and closed-form row max (rank-1 logits!)
    float an2 = av[b * NN + n0 + col] * LOG2E;
    float mx2 = fmaxf(an2 * bmm[b * 2 + 1], an2 * bmm[b * 2 + 0]);

    v8f acc0 = {}, acc1 = {}, acc2 = {}, acc3 = {}, acc4 = {};
    const half_t* fb  = fext + (size_t)b * CEXT * NN;
    const float*  bvb = bv + (size_t)b * NN;

    copy_stage(smem, 0, fb, bvb, 0);               // prime stage 0

    for (int ms = 0; ms < NN; ms += MSTEP) {
        int cur = (ms >> 7) & 1;
        stage_wait();
        __syncthreads();                           // stage `cur` visible to all waves
        if (ms + MSTEP < NN)
            copy_stage(smem, 1 - cur, fb, bvb, ms + MSTEP);  // overlap next copy

        const char*  stg  = smem + (size_t)cur * STAGEB;
        const float* sbst = (const float*)(stg + BOFFB);

#pragma unroll
        for (int s4 = 0; s4 < 4; ++s4) {
            int msub = s4 * 32;
            // --- P tile (16x32 f16) in A-matrix layout.
            // lane<16 holds K {0..15}, lane>=16 holds K {16..31}.
            const float4* bp = (const float4*)(sbst + msub + hi * 16);
            float4 q0 = bp[0], q1 = bp[1], q2 = bp[2], q3 = bp[3];
            float bk[16] = {q0.x, q0.y, q0.z, q0.w, q1.x, q1.y, q1.z, q1.w,
                            q2.x, q2.y, q2.z, q2.w, q3.x, q3.y, q3.z, q3.w};
            v16h aP;
#pragma unroll
            for (int e = 0; e < 16; ++e)
                aP[e] = (half_t)__builtin_amdgcn_exp2f(fmaf(an2, bk[e], -mx2));

            // --- B tiles from LDS: lane = channel column, 16 contiguous K halves.
            // Row pad (272B) keeps the 16 lanes on distinct bank quads.
#define LOAD_BT(T)                                                              \
            ({ const char* rb = stg + (size_t)((T) * 16 + col) * ROWB           \
                                    + (size_t)(msub + hi * 16) * 2;             \
               v8h lo = *(const v8h*)rb;                                        \
               v8h hh = *(const v8h*)(rb + 16);                                 \
               __builtin_shufflevector(lo, hh, 0, 1, 2, 3, 4, 5, 6, 7,          \
                                       8, 9, 10, 11, 12, 13, 14, 15); })
            v16h f0 = LOAD_BT(0);
            v16h f1 = LOAD_BT(1);
            v16h f2 = LOAD_BT(2);
            v16h f3 = LOAD_BT(3);
            v16h f4 = LOAD_BT(4);
#undef LOAD_BT

            acc0 = __builtin_amdgcn_wmma_f32_16x16x32_f16(false, aP, false, f0, (short)0, acc0, false, false);
            acc1 = __builtin_amdgcn_wmma_f32_16x16x32_f16(false, aP, false, f1, (short)0, acc1, false, false);
            acc2 = __builtin_amdgcn_wmma_f32_16x16x32_f16(false, aP, false, f2, (short)0, acc2, false, false);
            acc3 = __builtin_amdgcn_wmma_f32_16x16x32_f16(false, aP, false, f3, (short)0, acc3, false, false);
            acc4 = __builtin_amdgcn_wmma_f32_16x16x32_f16(false, aP, false, f4, (short)0, acc4, false, false);
        }
        __syncthreads();                           // all waves done reading `cur`
    }

    // Epilogue: D layout row M = r + 8*hi, col = lane&15. Row sums live in
    // acc4 column 0 -> lanes 0 (rows 0..7) and 16 (rows 8..15).
    const float* featb = feature + (size_t)b * CC * NN;
    float*       outb  = out     + (size_t)b * CC * NN;
    int srcl = lane & 16;
#pragma unroll
    for (int r = 0; r < 8; ++r) {
        float s   = __shfl(acc4[r], srcl, 32);
        float inv = __builtin_amdgcn_rcpf(s);
        int   n   = n0 + r + hi * 8;
        size_t i0 = (size_t)(col     ) * NN + n;
        size_t i1 = (size_t)(col + 16) * NN + n;
        size_t i2 = (size_t)(col + 32) * NN + n;
        size_t i3 = (size_t)(col + 48) * NN + n;
        outb[i0] = featb[i0] + acc0[r] * inv;
        outb[i1] = featb[i1] + acc1[r] * inv;
        outb[i2] = featb[i2] + acc2[r] * inv;
        outb[i3] = featb[i3] + acc3[r] * inv;
    }
}

// ---------------------------------------------------------------------- launch
extern "C" void kernel_launch(void* const* d_in, const int* in_sizes, int n_in,
                              void* d_out, int out_size, void* d_ws, size_t ws_size,
                              hipStream_t stream) {
    const float* x     = (const float*)d_in[0];
    const float* wa    = (const float*)d_in[1];
    const float* ba    = (const float*)d_in[2];
    const float* wb    = (const float*)d_in[3];
    const float* bb    = (const float*)d_in[4];
    const float* wm    = (const float*)d_in[5];
    const float* bm    = (const float*)d_in[6];
    const float* gamma = (const float*)d_in[7];
    const float* beta  = (const float*)d_in[8];
    const float* rmean = (const float*)d_in[9];
    const float* rvar  = (const float*)d_in[10];
    float* out = (float*)d_out;

    char* ws = (char*)d_ws;
    float*  av   = (float*)(ws);                                  // B*N f32
    float*  bv   = (float*)(ws + (size_t)BB * NN * 4);            // B*N f32
    float*  bmm  = (float*)(ws + (size_t)2 * BB * NN * 4);        // B*2 f32
    half_t* fext = (half_t*)(ws + (size_t)2 * BB * NN * 4 + 256); // B*80*N f16

    ab_kernel<<<dim3(NN / 256, BB), 256, 0, stream>>>(x, wa, ba, wb, bb, av, bv);
    bminmax_kernel<<<BB, 256, 0, stream>>>(bv, bmm);
    fext_kernel<<<BB * (NN / 128), 128, 0, stream>>>(x, wm, bm, gamma, beta,
                                                     rmean, rvar, fext);
    attn_kernel<<<BB * (NN / 64), 128, 0, stream>>>(x /*feature*/, av, bv, bmm,
                                                    fext, out);
}
